// ViT_38311108280540
// MI455X (gfx1250) — compile-verified, matrix operand order checked by reading, not run on hardware
//
#include <hip/hip_runtime.h>

#define L_ 12
#define E_ 768
#define M_ 3072
#define H_ 12
#define DH_ 64
#define PD_ 512
#define NC_ 1000
#define IMG_ 96
#define B_ 4
#define NPATCH_ 1728
#define S_ 1729
#define SFULL_ 1728   // S_ rounded down to key-tile (32) multiple

typedef _Float16 half_t;
typedef _Float16 v16h __attribute__((ext_vector_type(16)));
typedef _Float16 v8h  __attribute__((ext_vector_type(8)));
typedef _Float16 v4h  __attribute__((ext_vector_type(4)));
typedef float    v8f  __attribute__((ext_vector_type(8)));
typedef int      v4i  __attribute__((ext_vector_type(4)));

#define AS1 __attribute__((address_space(1)))
#define AS3 __attribute__((address_space(3)))

#if defined(__has_builtin)
#if __has_builtin(__builtin_amdgcn_global_load_async_to_lds_b128)
#define HAVE_ASYNC 1
#endif
#if __has_builtin(__builtin_amdgcn_ds_swizzle)
#define HAVE_SWZ 1
#endif
#endif

// Copy 16 bytes global -> LDS. Async (no VGPR round-trip, ASYNCcnt) when available.
__device__ __forceinline__ void copy16_g2l(void* lds, const void* g) {
#ifdef HAVE_ASYNC
    __builtin_amdgcn_global_load_async_to_lds_b128((AS1 v4i*)g, (AS3 v4i*)lds, 0, 0);
#else
    *(v8h*)lds = *(const v8h*)g;
#endif
}
__device__ __forceinline__ void wait_async0() {
#ifdef HAVE_ASYNC
    asm volatile("s_wait_asynccnt 0x0" ::: "memory");
#endif
}
__device__ __forceinline__ void wait_async2() {
#ifdef HAVE_ASYNC
    asm volatile("s_wait_asynccnt 0x2" ::: "memory");
#endif
}
__device__ __forceinline__ int imin(int a, int b) { return a < b ? a : b; }

// xor-lane exchange within 16-lane groups via ds_swizzle (immediate pattern),
// single LDS-unit op instead of the 5-op generic __shfl_xor lowering.
template <int XM>
__device__ __forceinline__ float swz_xor(float v) {
#ifdef HAVE_SWZ
    return __int_as_float(__builtin_amdgcn_ds_swizzle(__float_as_int(v), (XM << 10) | 0x1f));
#else
    return __shfl_xor(v, XM, 32);
#endif
}
__device__ __forceinline__ float red16_max(float v) {
    v = fmaxf(v, swz_xor<1>(v)); v = fmaxf(v, swz_xor<2>(v));
    v = fmaxf(v, swz_xor<4>(v)); v = fmaxf(v, swz_xor<8>(v));
    return v;
}
__device__ __forceinline__ float red16_sum(float v) {
    v += swz_xor<1>(v); v += swz_xor<2>(v);
    v += swz_xor<4>(v); v += swz_xor<8>(v);
    return v;
}

// Load a 16x32 f16 WMMA operand fragment from LDS (CDNA5 ISA 7.12.2 layout):
// lanes 0-15 -> row = lane, K chunks 0..7 & 16..23; lanes 16-31 -> row = lane-16,
// chunks 8..15 & 24..31. B operands use identical addressing for N-major tiles.
__device__ __forceinline__ v16h frag_ld(const half_t* base, int ldah) {
    const int l  = threadIdx.x & 31;
    const int r  = l & 15;
    const int kh = l >> 4;
    const half_t* p = base + (size_t)r * ldah;
    v8h lo = *(const v8h*)(p + kh * 8);
    v8h hi = *(const v8h*)(p + 16 + kh * 8);
    v16h f;
#pragma unroll
    for (int i = 0; i < 8; ++i) { f[i] = lo[i]; f[i + 8] = hi[i]; }
    return f;
}

// ---------------------------------------------------------------------------
// Tiled WMMA GEMM: C[M,N] = act(A[M,K](f16) * B[K,N](f32) + bias) + res
// 256 threads / 8 waves, tile 128x128, BK=32, double-buffered LDS with
// async-prefetch of the next K-tile. act: 0=none, 1=exact GELU.
// ---------------------------------------------------------------------------
template <typename CT>
__launch_bounds__(256)
__global__ void gemm_wmma(const half_t* __restrict__ A, const float* __restrict__ Bw,
                          const float* __restrict__ bias, const float* __restrict__ res,
                          CT* __restrict__ C, int Mr, int N, int K, int act)
{
    __shared__ __align__(16) half_t As[2][128 * 40];
    __shared__ __align__(16) half_t Bs[2][128 * 40];   // N-major: [n][k]

    const int t    = threadIdx.x;
    const int wave = t >> 5;
    const int wm   = wave >> 2;       // 0..1 : 64-row slab
    const int wn   = wave & 3;        // 0..3 : 32-col slab
    const int rowBase = blockIdx.y * 128;
    const int colBase = blockIdx.x * 128;

    // Per-thread staging geometry (loop-invariant)
    const int ar0 = t >> 1;                       // A: thread covers 2 chunks of row ar0
    const int aseg0 = (t & 1) * 2;                // chunk pair: segs {0,1} or {2,3}
    const half_t* arow = A + (size_t)imin(rowBase + ar0, Mr - 1) * K;
    const int bnq = t & 31, bkq = t >> 5;         // B: 4x4 micro-block
    const float* brow = Bw + (size_t)(bkq * 4) * N + colBase + bnq * 4;

    auto stage = [&](int kb, int buf) __attribute__((always_inline)) {
        // A tile: two 16B async chunks per thread
        copy16_g2l(&As[buf][ar0 * 40 + (aseg0 + 0) * 8], arow + kb + (aseg0 + 0) * 8);
        copy16_g2l(&As[buf][ar0 * 40 + (aseg0 + 1) * 8], arow + kb + (aseg0 + 1) * 8);
        // B tile transposed via 4x4 micro-block -> 4x ds_store_b64
        const float* src = brow + (size_t)kb * N;
        float4 r0 = *(const float4*)(src);
        float4 r1 = *(const float4*)(src + N);
        float4 r2 = *(const float4*)(src + 2 * N);
        float4 r3 = *(const float4*)(src + 3 * N);
        half_t* bp = &Bs[buf][(bnq * 4) * 40 + bkq * 4];
        v4h c0 = { (half_t)r0.x, (half_t)r1.x, (half_t)r2.x, (half_t)r3.x };
        v4h c1 = { (half_t)r0.y, (half_t)r1.y, (half_t)r2.y, (half_t)r3.y };
        v4h c2 = { (half_t)r0.z, (half_t)r1.z, (half_t)r2.z, (half_t)r3.z };
        v4h c3 = { (half_t)r0.w, (half_t)r1.w, (half_t)r2.w, (half_t)r3.w };
        *(v4h*)(bp)       = c0;
        *(v4h*)(bp + 40)  = c1;
        *(v4h*)(bp + 80)  = c2;
        *(v4h*)(bp + 120) = c3;
    };

    v8f acc[4][2] = {};
    const int nk = K >> 5;

    stage(0, 0);
    for (int k = 0; k < nk; ++k) {
        const int buf = k & 1;
        if (k + 1 < nk) { stage((k + 1) << 5, buf ^ 1); wait_async2(); }
        else           { wait_async0(); }
        __syncthreads();   // all waves' tile-k copies complete & visible

        v16h af[4], bf[2];
#pragma unroll
        for (int mt = 0; mt < 4; ++mt) af[mt] = frag_ld(&As[buf][(wm * 64 + mt * 16) * 40], 40);
#pragma unroll
        for (int nt = 0; nt < 2; ++nt) bf[nt] = frag_ld(&Bs[buf][(wn * 32 + nt * 16) * 40], 40);
#pragma unroll
        for (int mt = 0; mt < 4; ++mt)
#pragma unroll
            for (int nt = 0; nt < 2; ++nt)
                acc[mt][nt] = __builtin_amdgcn_wmma_f32_16x16x32_f16(
                    false, af[mt], false, bf[nt], (short)0, acc[mt][nt], false, false);
        __syncthreads();   // reads done before buf is overwritten next iteration
    }

    const int l  = t & 31;
    const int lr = l & 15;
    const int lh = l >> 4;
#pragma unroll
    for (int mt = 0; mt < 4; ++mt) {
#pragma unroll
        for (int nt = 0; nt < 2; ++nt) {
            int col = colBase + wn * 32 + nt * 16 + lr;
            float bv = bias ? bias[col] : 0.f;
#pragma unroll
            for (int i = 0; i < 8; ++i) {
                int row = rowBase + wm * 64 + mt * 16 + i + 8 * lh;
                if (row < Mr) {
                    float v = acc[mt][nt][i] + bv;
                    if (act == 1) v = 0.5f * v * (1.f + erff(v * 0.70710678118f));
                    if (res) v += res[(size_t)row * N + col];
                    C[(size_t)row * N + col] = (CT)v;
                }
            }
        }
    }
}

// ---------------------------------------------------------------------------
// Flash attention: WMMA for QK^T and PV. 128 queries/block (8 waves x 16 rows),
// one (batch, head) per blockIdx.y. qkv: f16 [B*S, 3E] (q | k | v, per head).
// Main loop runs unmasked full key-tiles; single masked tail tile at the end.
// ---------------------------------------------------------------------------
__launch_bounds__(256)
__global__ void attn_kernel(const half_t* __restrict__ qkv, half_t* __restrict__ outp)
{
    __shared__ __align__(16) half_t Ks[32 * 72];      // [key][dh]
    __shared__ __align__(16) half_t Vt[64 * 40];      // [dh][key]
    __shared__ __align__(16) half_t Ps[8 * 16 * 40];  // per-wave P tile

    const int t    = threadIdx.x;
    const int wave = t >> 5;
    const int l    = t & 31;
    const int lr   = l & 15;
    const int lh   = l >> 4;
    const int b    = blockIdx.y / H_;
    const int h    = blockIdx.y % H_;
    const int qbase = blockIdx.x * 128 + wave * 16;
    const float scale = 0.125f;   // DH^-0.5

    // Q fragments (A layout); clamp OOB rows (their outputs are never stored)
    v16h aq[2];
    {
        int q = imin(qbase + lr, S_ - 1);
        const half_t* qp = qkv + (size_t)(b * S_ + q) * (3 * E_) + h * DH_;
#pragma unroll
        for (int c = 0; c < 2; ++c) {
            v8h lo = *(const v8h*)(qp + c * 32 + lh * 8);
            v8h hi = *(const v8h*)(qp + c * 32 + 16 + lh * 8);
            v16h f;
#pragma unroll
            for (int j = 0; j < 8; ++j) { f[j] = lo[j]; f[j + 8] = hi[j]; }
            aq[c] = f;
        }
    }

    float mrow[8], lrow[8];
    v8f o[4] = {};
#pragma unroll
    for (int i = 0; i < 8; ++i) { mrow[i] = -1e30f; lrow[i] = 0.f; }

    auto process_tile = [&](int kb, bool tail) __attribute__((always_inline)) {
        __syncthreads();
        // K tile (32x64 halves): 256 16B chunks, async copy
        {
            int kr = t >> 3, seg = t & 7;
            int kg = tail ? imin(kb + kr, S_ - 1) : (kb + kr);
            copy16_g2l(&Ks[kr * 72 + seg * 8],
                       qkv + (size_t)(b * S_ + kg) * (3 * E_) + E_ + h * DH_ + seg * 8);
        }
        // V tile transposed: 4key x 2dh micro-blocks -> ds_store_b64
        {
            int kq = t & 7, dp = t >> 3;           // 8 key-quads x 32 dh-pairs
            half_t a0[4], a1[4];
#pragma unroll
            for (int j = 0; j < 4; ++j) {
                int kg = tail ? imin(kb + kq * 4 + j, S_ - 1) : (kb + kq * 4 + j);
                const half_t* vp = qkv + (size_t)(b * S_ + kg) * (3 * E_) + 2 * E_ + h * DH_ + dp * 2;
                a0[j] = vp[0]; a1[j] = vp[1];
            }
            v4h c0 = { a0[0], a0[1], a0[2], a0[3] };
            v4h c1 = { a1[0], a1[1], a1[2], a1[3] };
            *(v4h*)&Vt[(dp * 2 + 0) * 40 + kq * 4] = c0;
            *(v4h*)&Vt[(dp * 2 + 1) * 40 + kq * 4] = c1;
        }
        wait_async0();
        __syncthreads();

        // scores for two 16-key subtiles (DH=64 -> 2 chained WMMAs each)
        v8f sc[2];
#pragma unroll
        for (int n = 0; n < 2; ++n) {
            v8f s = {};
            v16h bk0 = frag_ld(&Ks[(n * 16) * 72 + 0], 72);
            v16h bk1 = frag_ld(&Ks[(n * 16) * 72 + 32], 72);
            s = __builtin_amdgcn_wmma_f32_16x16x32_f16(false, aq[0], false, bk0, (short)0, s, false, false);
            s = __builtin_amdgcn_wmma_f32_16x16x32_f16(false, aq[1], false, bk1, (short)0, s, false, false);
            sc[n] = s;
        }

        bool valid0 = !tail || (kb + lr) < S_;
        bool valid1 = !tail || (kb + 16 + lr) < S_;
#pragma unroll
        for (int i = 0; i < 8; ++i) {
            float s0 = valid0 ? sc[0][i] * scale : -1e30f;
            float s1 = valid1 ? sc[1][i] * scale : -1e30f;
            float m = red16_max(fmaxf(s0, s1));
            float newm = fmaxf(mrow[i], m);
            float fac  = __expf(mrow[i] - newm);
            float p0 = valid0 ? __expf(s0 - newm) : 0.f;
            float p1 = valid1 ? __expf(s1 - newm) : 0.f;
            float ts = red16_sum(p0 + p1);
            lrow[i] = lrow[i] * fac + ts;
            mrow[i] = newm;
#pragma unroll
            for (int nt = 0; nt < 4; ++nt) o[nt][i] *= fac;
            int prow = (wave * 16 + i + 8 * lh) * 40;
            Ps[prow + lr]      = (half_t)p0;
            Ps[prow + 16 + lr] = (half_t)p1;
        }
        __syncthreads();

        // O += P(16x32) x V(32x64): 4 WMMAs over dh tiles
        v16h ap = frag_ld(&Ps[(wave * 16) * 40], 40);
#pragma unroll
        for (int nt = 0; nt < 4; ++nt) {
            v16h bv = frag_ld(&Vt[(nt * 16) * 40], 40);
            o[nt] = __builtin_amdgcn_wmma_f32_16x16x32_f16(false, ap, false, bv, (short)0, o[nt], false, false);
        }
    };

    for (int kb = 0; kb < SFULL_; kb += 32) process_tile(kb, false);
    process_tile(SFULL_, true);     // S_ % 32 != 0 -> masked tail tile

    // merged-head f16 output [B*S, E]
#pragma unroll
    for (int i = 0; i < 8; ++i) {
        int q = qbase + i + 8 * lh;
        if (q < S_) {
            float inv = lrow[i] > 0.f ? 1.f / lrow[i] : 0.f;
#pragma unroll
            for (int nt = 0; nt < 4; ++nt)
                outp[(size_t)(b * S_ + q) * E_ + h * DH_ + nt * 16 + lr] = (half_t)(o[nt][i] * inv);
        }
    }
}

// ---------------------------------------------------------------------------
// LayerNorm (fp32 in -> f16 out), one block per row, configurable strides.
// ---------------------------------------------------------------------------
__launch_bounds__(256)
__global__ void ln_kernel(const float* __restrict__ in, const float* __restrict__ g,
                          const float* __restrict__ bta, half_t* __restrict__ out,
                          int len, long instride, long outstride)
{
    __shared__ float red[256];
    const float* x = in  + (long)blockIdx.x * instride;
    half_t*      o = out + (long)blockIdx.x * outstride;
    float s = 0.f;
    for (int i = threadIdx.x; i < len; i += 256) s += x[i];
    red[threadIdx.x] = s; __syncthreads();
    for (int off = 128; off > 0; off >>= 1) {
        if (threadIdx.x < off) red[threadIdx.x] += red[threadIdx.x + off];
        __syncthreads();
    }
    float mean = red[0] / len;
    __syncthreads();
    float v = 0.f;
    for (int i = threadIdx.x; i < len; i += 256) { float d = x[i] - mean; v += d * d; }
    red[threadIdx.x] = v; __syncthreads();
    for (int off = 128; off > 0; off >>= 1) {
        if (threadIdx.x < off) red[threadIdx.x] += red[threadIdx.x + off];
        __syncthreads();
    }
    float rstd = rsqrtf(red[0] / len + 1e-5f);
    for (int i = threadIdx.x; i < len; i += 256)
        o[i] = (half_t)((x[i] - mean) * rstd * g[i] + bta[i]);
}

// ---------------------------------------------------------------------------
// Patch gather (8x8x8, c=1) + LayerNorm(512) -> A0 f16 [B*NPATCH, 512]
// ---------------------------------------------------------------------------
__launch_bounds__(256)
__global__ void patch_ln1_kernel(const float* __restrict__ x, const float* __restrict__ g,
                                 const float* __restrict__ bt, half_t* __restrict__ A0)
{
    __shared__ float red[256];
    int pid = blockIdx.x;
    int b  = pid / NPATCH_;
    int p  = pid % NPATCH_;
    int pd = p / 144, ph = (p / 12) % 12, pw = p % 12;
    float vals[2]; int es[2];
#pragma unroll
    for (int u = 0; u < 2; ++u) {
        int e = threadIdx.x + u * 256;
        es[u] = e;
        int pz = e >> 6, py = (e >> 3) & 7, px = e & 7;
        size_t idx = (((size_t)b * IMG_ + (pd * 8 + pz)) * IMG_ + (ph * 8 + py)) * IMG_ + (pw * 8 + px);
        vals[u] = x[idx];
    }
    float s = vals[0] + vals[1];
    red[threadIdx.x] = s; __syncthreads();
    for (int off = 128; off > 0; off >>= 1) {
        if (threadIdx.x < off) red[threadIdx.x] += red[threadIdx.x + off];
        __syncthreads();
    }
    float mean = red[0] * (1.f / 512.f);
    __syncthreads();
    float v = 0.f;
#pragma unroll
    for (int u = 0; u < 2; ++u) { float d = vals[u] - mean; v += d * d; }
    red[threadIdx.x] = v; __syncthreads();
    for (int off = 128; off > 0; off >>= 1) {
        if (threadIdx.x < off) red[threadIdx.x] += red[threadIdx.x + off];
        __syncthreads();
    }
    float rstd = rsqrtf(red[0] * (1.f / 512.f) + 1e-5f);
#pragma unroll
    for (int u = 0; u < 2; ++u) {
        int e = es[u];
        A0[(size_t)pid * 512 + e] = (half_t)((vals[u] - mean) * rstd * g[e] + bt[e]);
    }
}

// concat cls + add positional embedding -> t fp32 [B, S, E]
__global__ void build_t(const half_t* __restrict__ proj, const float* __restrict__ cls,
                        const float* __restrict__ pos, float* __restrict__ tbuf)
{
    long i = (long)blockIdx.x * 256 + threadIdx.x;
    long total = (long)B_ * S_ * E_;
    if (i >= total) return;
    int  e  = (int)(i % E_);
    long bs = i / E_;
    int  s  = (int)(bs % S_);
    int  b  = (int)(bs / S_);
    float v = (s == 0) ? cls[e] : (float)proj[((long)b * NPATCH_ + (s - 1)) * E_ + e];
    tbuf[i] = v + pos[(long)s * E_ + e];
}

// tiny classifier head: out[b,c] = hn[b,:] . Wh[:,c] + bh[c]
__global__ void head_kernel(const half_t* __restrict__ hn, const float* __restrict__ Wh,
                            const float* __restrict__ bh, float* __restrict__ out)
{
    int i = blockIdx.x * 256 + threadIdx.x;
    if (i >= B_ * NC_) return;
    int b = i / NC_, c = i % NC_;
    float s = bh[c];
    for (int e = 0; e < E_; ++e) s += (float)hn[b * E_ + e] * Wh[(size_t)e * NC_ + c];
    out[i] = s;
}

// ---------------------------------------------------------------------------
extern "C" void kernel_launch(void* const* d_in, const int* in_sizes, int n_in,
                              void* d_out, int out_size, void* d_ws, size_t ws_size,
                              hipStream_t stream)
{
    (void)in_sizes; (void)n_in; (void)out_size; (void)ws_size;
    const float* x    = (const float*)d_in[0];
    const float* pl1g = (const float*)d_in[1];
    const float* pl1b = (const float*)d_in[2];
    const float* Wp   = (const float*)d_in[3];
    const float* bp   = (const float*)d_in[4];
    const float* pl2g = (const float*)d_in[5];
    const float* pl2b = (const float*)d_in[6];
    const float* pos  = (const float*)d_in[7];
    const float* cls  = (const float*)d_in[8];
    const float* alng = (const float*)d_in[9];
    const float* alnb = (const float*)d_in[10];
    const float* Wqkv = (const float*)d_in[11];
    const float* Wo   = (const float*)d_in[12];
    const float* bo   = (const float*)d_in[13];
    const float* flng = (const float*)d_in[14];
    const float* flnb = (const float*)d_in[15];
    const float* W1   = (const float*)d_in[16];
    const float* b1   = (const float*)d_in[17];
    const float* W2   = (const float*)d_in[18];
    const float* b2   = (const float*)d_in[19];
    const float* hlng = (const float*)d_in[20];
    const float* hlnb = (const float*)d_in[21];
    const float* Wh   = (const float*)d_in[22];
    const float* bh   = (const float*)d_in[23];
    float* out = (float*)d_out;

    float* ws = (float*)d_ws;
    const size_t tElems = (size_t)B_ * S_ * E_;           // 5,311,488
    float*  t   = ws;                                     // residual stream (fp32)
    half_t* xn  = (half_t*)(ws + tElems);                 // f16: LN out / attn out / head scratch
    float*  big = ws + tElems + tElems / 2;               // shared scratch region
    half_t* A0   = (half_t*)big;                          // 6912*512 halves
    float*  tmp  = big + 2 * 1024 * 1024;                 // 6912*768 fp32 (disjoint from A0)
    half_t* qkvb = (half_t*)big;                          // 6916*2304 halves
    half_t* ffb  = (half_t*)big;                          // 6916*3072 halves
    half_t* attno = xn;                                   // reuse after qkv GEMM consumed xn

    const int rows = B_ * S_;                             // 6916
    dim3 blk(256);

    // ---- patch embedding ----
    patch_ln1_kernel<<<B_ * NPATCH_, blk, 0, stream>>>(x, pl1g, pl1b, A0);
    gemm_wmma<float><<<dim3(E_ / 128, (B_ * NPATCH_ + 127) / 128), blk, 0, stream>>>(
        A0, Wp, bp, nullptr, tmp, B_ * NPATCH_, E_, PD_, 0);
    ln_kernel<<<B_ * NPATCH_, blk, 0, stream>>>(tmp, pl2g, pl2b, xn, E_, E_, E_);
    {
        long total = (long)B_ * S_ * E_;
        build_t<<<(unsigned)((total + 255) / 256), blk, 0, stream>>>(xn, cls, pos, t);
    }

    // ---- transformer layers ----
    for (int l = 0; l < L_; ++l) {
        ln_kernel<<<rows, blk, 0, stream>>>(t, alng + (size_t)l * E_, alnb + (size_t)l * E_, xn, E_, E_, E_);
        gemm_wmma<half_t><<<dim3((3 * E_) / 128, (rows + 127) / 128), blk, 0, stream>>>(
            xn, Wqkv + (size_t)l * E_ * 3 * E_, nullptr, nullptr, qkvb, rows, 3 * E_, E_, 0);
        attn_kernel<<<dim3((S_ + 127) / 128, B_ * H_), blk, 0, stream>>>(qkvb, attno);
        gemm_wmma<float><<<dim3(E_ / 128, (rows + 127) / 128), blk, 0, stream>>>(
            attno, Wo + (size_t)l * E_ * E_, bo + (size_t)l * E_, t, t, rows, E_, E_, 0);
        ln_kernel<<<rows, blk, 0, stream>>>(t, flng + (size_t)l * E_, flnb + (size_t)l * E_, xn, E_, E_, E_);
        gemm_wmma<half_t><<<dim3(M_ / 128, (rows + 127) / 128), blk, 0, stream>>>(
            xn, W1 + (size_t)l * E_ * M_, b1 + (size_t)l * M_, nullptr, ffb, rows, M_, E_, 1);
        gemm_wmma<float><<<dim3(E_ / 128, (rows + 127) / 128), blk, 0, stream>>>(
            ffb, W2 + (size_t)l * M_ * E_, b2 + (size_t)l * E_, t, t, rows, E_, M_, 0);
    }

    // ---- head: LN on cls rows (stride S*E), then 4x1000 matvec ----
    ln_kernel<<<B_, blk, 0, stream>>>(t, hlng, hlnb, xn, E_, (long)S_ * E_, E_);
    head_kernel<<<(B_ * NC_ + 255) / 256, blk, 0, stream>>>(xn, Wh, bh, out);
}
